// mIoU_13967233646797
// MI455X (gfx1250) — compile-verified, matrix operand order checked by reading, not run on hardware
//
#include <hip/hip_runtime.h>

#define NBINS 91
#define NBP   92          // padded bin count
#define NREP  8           // LDS histogram replicas (lane % 8)
#define HBLK  256
#define HGRID 2048

typedef float __attribute__((ext_vector_type(4))) f32x4;
typedef int   __attribute__((ext_vector_type(4))) i32x4;
typedef __attribute__((ext_vector_type(16))) _Float16 v16h;
typedef __attribute__((ext_vector_type(8)))  float    v8f;

// ---------------- kernel 1: zero the global counters in d_ws ----------------
__global__ void miou_zero(unsigned* __restrict__ g) {
  int i = threadIdx.x;
  if (i < 3 * NBP) g[i] = 0u;
}

// ---------------- kernel 2: streaming 3-way histogram -----------------------
__global__ __launch_bounds__(HBLK)
void miou_hist(const f32x4* __restrict__ inp, const i32x4* __restrict__ tgt,
               unsigned* __restrict__ g, int nvec, int ntail,
               const float* __restrict__ tail_in, const int* __restrict__ tail_tg) {
  __shared__ unsigned h[NREP * 3 * NBP];
  for (int i = threadIdx.x; i < NREP * 3 * NBP; i += HBLK) h[i] = 0u;
  __syncthreads();

  unsigned* hin = &h[(threadIdx.x & (NREP - 1)) * 3 * NBP];
  unsigned* htg = hin + NBP;
  unsigned* hit = htg + NBP;

  const int stride = HGRID * HBLK;
  for (int i = blockIdx.x * HBLK + threadIdx.x; i < nvec; i += stride) {
    // speculative prefetch of next grid-stride iteration (global_prefetch_b8)
    __builtin_prefetch((const char*)(inp + i) + (size_t)stride * sizeof(f32x4), 0, 0);
    __builtin_prefetch((const char*)(tgt + i) + (size_t)stride * sizeof(i32x4), 0, 0);
    // single-pass streaming data: non-temporal b128 loads
    f32x4 f = __builtin_nontemporal_load(&inp[i]);
    i32x4 t = __builtin_nontemporal_load(&tgt[i]);
    int a0 = (int)f.x, a1 = (int)f.y, a2 = (int)f.z, a3 = (int)f.w;
    atomicAdd(&hin[a0], 1u); atomicAdd(&htg[t.x], 1u); if (a0 == t.x) atomicAdd(&hit[a0], 1u);
    atomicAdd(&hin[a1], 1u); atomicAdd(&htg[t.y], 1u); if (a1 == t.y) atomicAdd(&hit[a1], 1u);
    atomicAdd(&hin[a2], 1u); atomicAdd(&htg[t.z], 1u); if (a2 == t.z) atomicAdd(&hit[a2], 1u);
    atomicAdd(&hin[a3], 1u); atomicAdd(&htg[t.w], 1u); if (a3 == t.w) atomicAdd(&hit[a3], 1u);
  }

  // scalar tail (robustness; N here is divisible by 4)
  int gt = blockIdx.x * HBLK + threadIdx.x;
  if (gt < ntail) {
    int base = nvec * 4 + gt;
    int a = (int)tail_in[base];
    int b = tail_tg[base];
    atomicAdd(&hin[a], 1u); atomicAdd(&htg[b], 1u); if (a == b) atomicAdd(&hit[a], 1u);
  }

  __syncthreads();
  // fold replicas and merge into global counters
  for (int b = threadIdx.x; b < 3 * NBP; b += HBLK) {
    unsigned s = 0;
    for (int r = 0; r < NREP; ++r) s += h[r * 3 * NBP + b];
    if (s) atomicAdd(&g[b], s);
  }
}

// ------------- kernel 3: IoU + WMMA reduction (one wave32) -------------------
__global__ void miou_final(const unsigned* __restrict__ g,
                           const int* __restrict__ smooth_p,
                           float* __restrict__ out) {
  __shared__ float iou[NBINS];
  __shared__ float partial[32];
  const int lane = threadIdx.x;           // block of exactly 32 threads
  const float sm = (float)smooth_p[0];

  for (int i = lane; i < NBINS; i += 32) {
    float cin = (float)g[i];
    float ctg = (float)g[NBP + i];
    float itr = (float)g[2 * NBP + i];
    float uni = cin + ctg - itr;
    iou[i] = (i == 0) ? 0.0f : ((itr + sm) / (uni + sm));   // classes 1..90
  }
  __syncthreads();

  // A holds hi parts (slots 0..89) and f16 residual lo parts (slots 90..179)
  // of the 90 IoU values; B = all ones  =>  D row-sums; Sum(D) = 16 * total.
  union { v16h v; _Float16 e[16]; } ua, ub;
  for (int t = 0; t < 16; ++t) {
    int s = lane * 16 + t;                   // global slot 0..511
    int j = (s < 90) ? (s + 1) : (s - 89);   // class id for hi / lo region
    j = (j < NBINS) ? j : 0;                 // clamp (iou[0] == 0)
    float v = iou[j];
    _Float16 hi = (_Float16)v;
    _Float16 lo = (_Float16)(v - (float)hi);
    _Float16 e  = (s < 90) ? hi : lo;
    ua.e[t] = (s < 180) ? e : (_Float16)0.0f;
    ub.e[t] = (_Float16)1.0f;
  }

  v8f c = {};
  c = __builtin_amdgcn_wmma_f32_16x16x32_f16(
      /*neg_a=*/false, ua.v, /*neg_b=*/false, ub.v,
      /*c_mod=*/(short)0, c, /*reuse_a=*/false, /*reuse_b=*/false);

  union { v8f v; float e[8]; } uc; uc.v = c;
  float ps = 0.0f;
  for (int k = 0; k < 8; ++k) ps += uc.e[k];
  partial[lane] = ps;
  __syncthreads();
  if (lane == 0) {
    float tot = 0.0f;
    for (int k = 0; k < 32; ++k) tot += partial[k];
    tot *= (1.0f / 16.0f);                  // every column repeats the row-sums
    out[0] = tot * (1.0f / 90.0f);          // mean over classes 1..90
  }
}

// ----------------------------- launcher -------------------------------------
extern "C" void kernel_launch(void* const* d_in, const int* in_sizes, int n_in,
                              void* d_out, int out_size, void* d_ws, size_t ws_size,
                              hipStream_t stream) {
  const float* inp    = (const float*)d_in[0];
  const int*   tgt    = (const int*)d_in[1];
  const int*   smooth = (const int*)d_in[2];
  unsigned*    g      = (unsigned*)d_ws;          // 3*92 u32 counters

  int n     = in_sizes[0];
  int nvec  = n >> 2;
  int ntail = n & 3;

  miou_zero<<<1, 512, 0, stream>>>(g);
  miou_hist<<<HGRID, HBLK, 0, stream>>>((const f32x4*)inp, (const i32x4*)tgt,
                                        g, nvec, ntail, inp, tgt);
  miou_final<<<1, 32, 0, stream>>>(g, smooth, (float*)d_out);
}
